// ThermalAttention_25958782337785
// MI455X (gfx1250) — matrix operand order repaired and checked by reading, not compiled, not run on hardware
//
#include <hip/hip_runtime.h>

// ---------- types ----------
typedef __bf16 bf16;
typedef __attribute__((ext_vector_type(16))) __bf16 v16bf;
typedef __attribute__((ext_vector_type(8)))  __bf16 v8bf;
typedef __attribute__((ext_vector_type(8)))  float  v8f;
typedef __attribute__((ext_vector_type(4)))  unsigned int u32x4;
typedef __attribute__((ext_vector_type(8)))  unsigned int u32x8;

union Frag { v16bf v; v8bf h[2]; };

__device__ inline v8f vzero8() { v8f z; for (int i = 0; i < 8; ++i) z[i] = 0.f; return z; }

__device__ inline v8f wmma_bf16(const Frag& a, const Frag& b, v8f c) {
    return __builtin_amdgcn_wmma_f32_16x16x32_bf16(false, a.v, false, b.v,
                                                   (short)0, c, false, false);
}

// ---------- Tensor Data Mover: 2D bf16 tile with LDS row padding ----------
// D# group0: [1:0]=count=1, [63:32]=lds_addr, [120:64]=global_addr, [127:126]=type=2
// D# group1: [17:16]=data_size(1->2B), [20]=pad_enable, [24:22]=pad_interval code,
//            [31:25]=pad_amount code, [79:48]=tensor_dim0, [111:80]=tensor_dim1,
//            [127:112]=tile_dim0, [143:128]=tile_dim1, [207:160]=tensor_dim0_stride
// pad_interval code: 3 -> every 16 DWORDs, 4 -> every 32 DWORDs.
// pad_amount   code: 3 -> 4 DWORDs (16B, keeps b128 alignment).
__device__ inline void tdm_load_2d_bf16(const bf16* gptr, unsigned lds_off,
                                        unsigned d0, unsigned d1, unsigned stride,
                                        unsigned pad_int_code) {
    unsigned long long ga = (unsigned long long)(uintptr_t)gptr;
    u32x4 g0;
    g0[0] = 1u;                                                   // count = 1
    g0[1] = lds_off;
    g0[2] = (unsigned)ga;
    g0[3] = ((unsigned)(ga >> 32) & 0x01FFFFFFu) | (2u << 30);    // type = 2 (image)
    u32x8 g1;
    g1[0] = (1u << 16)                                            // data_size = 2 bytes
          | (1u << 20)                                            // pad_enable
          | (pad_int_code << 22)                                  // pad every row
          | (3u << 25);                                           // pad 4 DWORDs
    g1[1] = (d0 & 0xFFFFu) << 16;                                 // tensor_dim0 lo
    g1[2] = ((d0 >> 16) & 0xFFFFu) | ((d1 & 0xFFFFu) << 16);      // dim0 hi | dim1 lo
    g1[3] = ((d1 >> 16) & 0xFFFFu) | ((d0 & 0xFFFFu) << 16);      // dim1 hi | tile_dim0
    g1[4] = d1 & 0xFFFFu;                                         // tile_dim1 (tile_dim2=0)
    g1[5] = stride;                                               // tensor_dim0_stride lo
    g1[6] = 0u;
    g1[7] = 0u;
    asm volatile("tensor_load_to_lds %0, %1" :: "s"(g0), "s"(g1) : "memory");
}

#define D_MODEL 1024
#define N3      3072
#define HEADS   16
#define HD      64
#define SEQ     2048
#define BS      8192   // B*S
#define GP      40     // GEMM LDS row pitch (32 data + 8 pad bf16) = 80B, conflict-free
#define KP      72     // attn LDS row pitch (64 data + 8 pad bf16) = 144B, conflict-free

// =====================================================================
// Pre-pass A: fp32 -> bf16 elementwise (x).
// =====================================================================
__global__ __launch_bounds__(256)
void cvt_bf16_kernel(const float* __restrict__ src, bf16* __restrict__ dst)
{
    size_t i = ((size_t)blockIdx.x * 256 + threadIdx.x) * 8;
    float4 f0 = *(const float4*)&src[i];
    float4 f1 = *(const float4*)&src[i + 4];
    v8bf o;
    o[0] = (bf16)f0.x; o[1] = (bf16)f0.y; o[2] = (bf16)f0.z; o[3] = (bf16)f0.w;
    o[4] = (bf16)f1.x; o[5] = (bf16)f1.y; o[6] = (bf16)f1.z; o[7] = (bf16)f1.w;
    *(v8bf*)&dst[i] = o;
}

// =====================================================================
// Pre-pass B: fp32 [R][C] -> bf16 transposed [C][R], 32x32 LDS tiles.
// =====================================================================
__global__ __launch_bounds__(256)
void transpose_cvt_kernel(const float* __restrict__ src, bf16* __restrict__ dst,
                          int R, int C)
{
    __shared__ bf16 tile[32][33];
    const int c0 = blockIdx.x * 32, r0 = blockIdx.y * 32;
    const int tr = threadIdx.x >> 5, tc = threadIdx.x & 31;
#pragma unroll
    for (int i = 0; i < 4; ++i) {
        int r = tr + i * 8;
        tile[r][tc] = (bf16)src[(size_t)(r0 + r) * C + c0 + tc];
    }
    __syncthreads();
#pragma unroll
    for (int i = 0; i < 4; ++i) {
        int c = tr + i * 8;
        dst[(size_t)(c0 + c) * R + r0 + tc] = tile[tc][c];
    }
}

// =====================================================================
// Kernel 1: qkv = xb @ wT^T (bf16, TDM-staged w/ pad, double-buffered);
// fused per-head LayerNorm on Q,K. Q,K out [B,H,S,Dh]; V out [B,H,Dh,S].
// =====================================================================
__global__ __launch_bounds__(256)
void qkv_ln_kernel(const bf16* __restrict__ xb, const bf16* __restrict__ wT,
                   const float* __restrict__ qgam, const float* __restrict__ qbet,
                   const float* __restrict__ kgam, const float* __restrict__ kbet,
                   bf16* __restrict__ qo, bf16* __restrict__ ko, bf16* __restrict__ vo)
{
    __shared__ __attribute__((aligned(16))) bf16 As[2][128 * GP];
    __shared__ __attribute__((aligned(16))) bf16 Bt[2][128 * GP];

    const int t = threadIdx.x;
    const int wid = t >> 5, lane = t & 31;
    const int lrow = lane & 15, kgd = lane >> 4;
    const int m0 = blockIdx.x * 128;
    const int n0 = blockIdx.y * 128;
    const int wm = (wid >> 1) * 32;
    const int wn = (wid & 1) * 64;
    const unsigned as_lds[2] = { (unsigned)(uintptr_t)&As[0][0],
                                 (unsigned)(uintptr_t)&As[1][0] };
    const unsigned bt_lds[2] = { (unsigned)(uintptr_t)&Bt[0][0],
                                 (unsigned)(uintptr_t)&Bt[1][0] };

    v8f acc[2][4];
    for (int i = 0; i < 2; ++i) for (int j = 0; j < 4; ++j) acc[i][j] = vzero8();

    if (wid == 0) {   // prologue DMA: tile 0
        tdm_load_2d_bf16(&xb[(size_t)m0 * D_MODEL], as_lds[0], 32, 128, D_MODEL, 3);
        tdm_load_2d_bf16(&wT[(size_t)n0 * D_MODEL], bt_lds[0], 32, 128, D_MODEL, 3);
        __builtin_amdgcn_s_wait_tensorcnt(0);
    }
    __syncthreads();

    int buf = 0;
    for (int k0 = 0; k0 < D_MODEL; k0 += 32, buf ^= 1) {
        if (wid == 0 && k0 + 32 < D_MODEL) {        // DMA next tile while computing
            tdm_load_2d_bf16(&xb[(size_t)m0 * D_MODEL + k0 + 32], as_lds[buf ^ 1], 32, 128, D_MODEL, 3);
            tdm_load_2d_bf16(&wT[(size_t)n0 * D_MODEL + k0 + 32], bt_lds[buf ^ 1], 32, 128, D_MODEL, 3);
        }
        Frag a[2], b[4];
#pragma unroll
        for (int rt = 0; rt < 2; ++rt) {
            const bf16* p = &As[buf][(wm + rt * 16 + lrow) * GP + kgd * 8];
            a[rt].h[0] = *(const v8bf*)p;
            a[rt].h[1] = *(const v8bf*)(p + 16);
        }
#pragma unroll
        for (int ct = 0; ct < 4; ++ct) {
            const bf16* p = &Bt[buf][(wn + ct * 16 + lrow) * GP + kgd * 8];
            b[ct].h[0] = *(const v8bf*)p;
            b[ct].h[1] = *(const v8bf*)(p + 16);
        }
#pragma unroll
        for (int rt = 0; rt < 2; ++rt)
#pragma unroll
            for (int ct = 0; ct < 4; ++ct)
                acc[rt][ct] = wmma_bf16(a[rt], b[ct], acc[rt][ct]);

        if (wid == 0) __builtin_amdgcn_s_wait_tensorcnt(0);
        __syncthreads();
    }

    // ---------- epilogue: per-head LayerNorm (Q,K) / transposed store (V) ----------
    const int wng    = n0 + wn;
    const int region = wng >> 10;          // 0=Q 1=K 2=V
    const int head   = (wng & 1023) >> 6;

    for (int rt = 0; rt < 2; ++rt) {
        if (region < 2) {
            float s1[8], s2[8];
#pragma unroll
            for (int v = 0; v < 8; ++v) {
                float a0 = 0.f, a1 = 0.f;
                for (int ct = 0; ct < 4; ++ct) {
                    float e = acc[rt][ct][v];
                    a0 += e; a1 += e * e;
                }
                for (int m = 1; m < 16; m <<= 1) {
                    a0 += __shfl_xor(a0, m, 32);
                    a1 += __shfl_xor(a1, m, 32);
                }
                s1[v] = a0; s2[v] = a1;
            }
            const float* gam = (region == 0) ? qgam : kgam;
            const float* bet = (region == 0) ? qbet : kbet;
            bf16* dst        = (region == 0) ? qo   : ko;
            for (int ct = 0; ct < 4; ++ct) {
                int col = ct * 16 + lrow;
                float g = gam[col], be = bet[col];
#pragma unroll
                for (int v = 0; v < 8; ++v) {
                    float mean = s1[v] * (1.f / 64.f);
                    float var  = s2[v] * (1.f / 64.f) - mean * mean;
                    float rstd = rsqrtf(var + 1e-5f);
                    float val  = (acc[rt][ct][v] - mean) * rstd * g + be;
                    int gm   = m0 + wm + rt * 16 + kgd * 8 + v;
                    int bidx = gm >> 11, srow = gm & 2047;
                    size_t off = (((size_t)(bidx * HEADS + head)) * SEQ + srow) * HD + col;
                    dst[off] = (bf16)val;
                }
            }
        } else {
            // V stored transposed: [B,H,Dh,S]
            for (int ct = 0; ct < 4; ++ct) {
                int col = ct * 16 + lrow;
#pragma unroll
                for (int v = 0; v < 8; ++v) {
                    int gm   = m0 + wm + rt * 16 + kgd * 8 + v;
                    int bidx = gm >> 11, srow = gm & 2047;
                    size_t off = (((size_t)(bidx * HEADS + head)) * HD + col) * SEQ + srow;
                    vo[off] = (bf16)acc[rt][ct][v];
                }
            }
        }
    }
}

// =====================================================================
// Kernel 2: flash attention per (b,h). K tile and V^T tile staged by the
// TDM with row padding, double-buffered against the WMMA pipeline.
// =====================================================================
__global__ __launch_bounds__(256)
void attn_kernel(const bf16* __restrict__ Q, const bf16* __restrict__ K,
                 const bf16* __restrict__ V, bf16* __restrict__ O)
{
    __shared__ __attribute__((aligned(16))) bf16 Ks[2][64 * KP];   // [key][d]
    __shared__ __attribute__((aligned(16))) bf16 Vt[2][64 * KP];   // [d][key]
    __shared__ __attribute__((aligned(16))) bf16 Ps[8][32 * KP];   // per-wave P tile

    const int t = threadIdx.x, wid = t >> 5, lane = t & 31;
    const int lrow = lane & 15, kgd = lane >> 4;
    const int bh = blockIdx.y;
    const size_t base  = (size_t)bh * SEQ * HD;   // Q,K layout [B,H,S,Dh]
    const size_t vbase = (size_t)bh * HD * SEQ;   // V layout   [B,H,Dh,S]
    const int qm = blockIdx.x * 256 + wid * 32;
    const float scale = 0.125f;   // 1/sqrt(64), T=1
    const unsigned ks_lds[2] = { (unsigned)(uintptr_t)&Ks[0][0],
                                 (unsigned)(uintptr_t)&Ks[1][0] };
    const unsigned vt_lds[2] = { (unsigned)(uintptr_t)&Vt[0][0],
                                 (unsigned)(uintptr_t)&Vt[1][0] };

    Frag qf[2][2];
#pragma unroll
    for (int rt = 0; rt < 2; ++rt)
#pragma unroll
        for (int kt = 0; kt < 2; ++kt) {
            const bf16* p = &Q[base + (size_t)(qm + rt * 16 + lrow) * HD + kt * 32 + kgd * 8];
            qf[rt][kt].h[0] = *(const v8bf*)p;
            qf[rt][kt].h[1] = *(const v8bf*)(p + 16);
        }

    v8f oacc[2][4];
    for (int i = 0; i < 2; ++i) for (int j = 0; j < 4; ++j) oacc[i][j] = vzero8();
    float rm[2][8], rl[2][8];
    for (int i = 0; i < 2; ++i) for (int v = 0; v < 8; ++v) { rm[i][v] = -3.0e38f; rl[i][v] = 0.f; }

    if (wid == 0) {   // prologue DMA: key tile 0
        tdm_load_2d_bf16(&K[base],  ks_lds[0], HD, 64, HD,  4);
        tdm_load_2d_bf16(&V[vbase], vt_lds[0], 64, HD, SEQ, 4);
        __builtin_amdgcn_s_wait_tensorcnt(0);
    }
    __syncthreads();

    int buf = 0;
    for (int j0 = 0; j0 < SEQ; j0 += 64, buf ^= 1) {
        if (wid == 0 && j0 + 64 < SEQ) {   // DMA next key tile while computing
            tdm_load_2d_bf16(&K[base + (size_t)(j0 + 64) * HD], ks_lds[buf ^ 1], HD, 64, HD,  4);
            tdm_load_2d_bf16(&V[vbase + j0 + 64],               vt_lds[buf ^ 1], 64, HD, SEQ, 4);
        }

        // ---- S = Q K^T (fp32 accum) ----
        v8f sacc[2][4];
        for (int i = 0; i < 2; ++i) for (int j = 0; j < 4; ++j) sacc[i][j] = vzero8();
#pragma unroll
        for (int kt = 0; kt < 2; ++kt) {
            Frag kb[4];
#pragma unroll
            for (int ct = 0; ct < 4; ++ct) {
                const bf16* p = &Ks[buf][(ct * 16 + lrow) * KP + kt * 32 + kgd * 8];
                kb[ct].h[0] = *(const v8bf*)p;
                kb[ct].h[1] = *(const v8bf*)(p + 16);
            }
#pragma unroll
            for (int rt = 0; rt < 2; ++rt)
#pragma unroll
                for (int ct = 0; ct < 4; ++ct)
                    sacc[rt][ct] = wmma_bf16(qf[rt][kt], kb[ct], sacc[rt][ct]);
        }

        // ---- online softmax; write P (bf16) to per-wave padded LDS ----
#pragma unroll
        for (int rt = 0; rt < 2; ++rt) {
#pragma unroll
            for (int v = 0; v < 8; ++v) {
                float mx = -3.0e38f;
                for (int ct = 0; ct < 4; ++ct) mx = fmaxf(mx, sacc[rt][ct][v]);
                for (int m = 1; m < 16; m <<= 1) mx = fmaxf(mx, __shfl_xor(mx, m, 32));
                mx *= scale;
                float mnew = fmaxf(rm[rt][v], mx);
                float f = __expf(rm[rt][v] - mnew);
                rm[rt][v] = mnew;
                int row = rt * 16 + kgd * 8 + v;
                float psum = 0.f;
                for (int ct = 0; ct < 4; ++ct) {
                    float p = __expf(sacc[rt][ct][v] * scale - mnew);
                    psum += p;
                    Ps[wid][row * KP + ct * 16 + lrow] = (bf16)p;
                }
                for (int m = 1; m < 16; m <<= 1) psum += __shfl_xor(psum, m, 32);
                rl[rt][v] = rl[rt][v] * f + psum;
                for (int ct = 0; ct < 4; ++ct) oacc[rt][ct][v] *= f;
            }
        }

        // ---- O += P * V ----
#pragma unroll
        for (int kt = 0; kt < 2; ++kt) {
            Frag pa[2], vb[4];
#pragma unroll
            for (int rt = 0; rt < 2; ++rt) {
                const bf16* p = &Ps[wid][(rt * 16 + lrow) * KP + kt * 32 + kgd * 8];
                pa[rt].h[0] = *(const v8bf*)p;
                pa[rt].h[1] = *(const v8bf*)(p + 16);
            }
#pragma unroll
            for (int ct = 0; ct < 4; ++ct) {
                const bf16* p = &Vt[buf][(ct * 16 + lrow) * KP + kt * 32 + kgd * 8];
                vb[ct].h[0] = *(const v8bf*)p;
                vb[ct].h[1] = *(const v8bf*)(p + 16);
            }
#pragma unroll
            for (int rt = 0; rt < 2; ++rt)
#pragma unroll
                for (int ct = 0; ct < 4; ++ct)
                    oacc[rt][ct] = wmma_bf16(pa[rt], vb[ct], oacc[rt][ct]);
        }

        if (wid == 0) __builtin_amdgcn_s_wait_tensorcnt(0);
        __syncthreads();
    }

    // ---- epilogue: normalize, store bf16 attn-out [B,S,D] ----
    const int b = bh >> 4, hh = bh & 15;
    for (int rt = 0; rt < 2; ++rt)
        for (int ct = 0; ct < 4; ++ct)
#pragma unroll
            for (int v = 0; v < 8; ++v) {
                int s = qm + rt * 16 + kgd * 8 + v;
                float o = oacc[rt][ct][v] / rl[rt][v];
                O[((size_t)(b * SEQ + s)) * D_MODEL + hh * HD + ct * 16 + lrow] = (bf16)o;
            }
}

// =====================================================================
// Kernel 3: out = attn_out(bf16) @ w_projT^T (bf16), fp32 store.
// Both tiles TDM-staged with pad, double-buffered.
// =====================================================================
__global__ __launch_bounds__(256)
void proj_kernel(const bf16* __restrict__ A, const bf16* __restrict__ WT,
                 float* __restrict__ C)
{
    __shared__ __attribute__((aligned(16))) bf16 As[2][128 * GP];
    __shared__ __attribute__((aligned(16))) bf16 Bt[2][128 * GP];

    const int t = threadIdx.x;
    const int wid = t >> 5, lane = t & 31;
    const int lrow = lane & 15, kgd = lane >> 4;
    const int m0 = blockIdx.x * 128;
    const int n0 = blockIdx.y * 128;
    const int wm = (wid >> 1) * 32;
    const int wn = (wid & 1) * 64;
    const unsigned as_lds[2] = { (unsigned)(uintptr_t)&As[0][0],
                                 (unsigned)(uintptr_t)&As[1][0] };
    const unsigned bt_lds[2] = { (unsigned)(uintptr_t)&Bt[0][0],
                                 (unsigned)(uintptr_t)&Bt[1][0] };

    v8f acc[2][4];
    for (int i = 0; i < 2; ++i) for (int j = 0; j < 4; ++j) acc[i][j] = vzero8();

    if (wid == 0) {
        tdm_load_2d_bf16(&A[(size_t)m0 * D_MODEL],  as_lds[0], 32, 128, D_MODEL, 3);
        tdm_load_2d_bf16(&WT[(size_t)n0 * D_MODEL], bt_lds[0], 32, 128, D_MODEL, 3);
        __builtin_amdgcn_s_wait_tensorcnt(0);
    }
    __syncthreads();

    int buf = 0;
    for (int k0 = 0; k0 < D_MODEL; k0 += 32, buf ^= 1) {
        if (wid == 0 && k0 + 32 < D_MODEL) {
            tdm_load_2d_bf16(&A[(size_t)m0 * D_MODEL + k0 + 32],  as_lds[buf ^ 1], 32, 128, D_MODEL, 3);
            tdm_load_2d_bf16(&WT[(size_t)n0 * D_MODEL + k0 + 32], bt_lds[buf ^ 1], 32, 128, D_MODEL, 3);
        }
        Frag a[2], b[4];
#pragma unroll
        for (int rt = 0; rt < 2; ++rt) {
            const bf16* p = &As[buf][(wm + rt * 16 + lrow) * GP + kgd * 8];
            a[rt].h[0] = *(const v8bf*)p;
            a[rt].h[1] = *(const v8bf*)(p + 16);
        }
#pragma unroll
        for (int ct = 0; ct < 4; ++ct) {
            const bf16* p = &Bt[buf][(wn + ct * 16 + lrow) * GP + kgd * 8];
            b[ct].h[0] = *(const v8bf*)p;
            b[ct].h[1] = *(const v8bf*)(p + 16);
        }
#pragma unroll
        for (int rt = 0; rt < 2; ++rt)
#pragma unroll
            for (int ct = 0; ct < 4; ++ct)
                acc[rt][ct] = wmma_bf16(a[rt], b[ct], acc[rt][ct]);

        if (wid == 0) __builtin_amdgcn_s_wait_tensorcnt(0);
        __syncthreads();
    }

    for (int rt = 0; rt < 2; ++rt)
        for (int ct = 0; ct < 4; ++ct)
#pragma unroll
            for (int v = 0; v < 8; ++v) {
                int gm = m0 + wm + rt * 16 + kgd * 8 + v;
                int gn = n0 + wn + ct * 16 + lrow;
                C[(size_t)gm * D_MODEL + gn] = acc[rt][ct][v];
            }
}

// =====================================================================
extern "C" void kernel_launch(void* const* d_in, const int* in_sizes, int n_in,
                              void* d_out, int out_size, void* d_ws, size_t ws_size,
                              hipStream_t stream)
{
    const float* x      = (const float*)d_in[0];
    const float* w_qkv  = (const float*)d_in[1];
    const float* w_proj = (const float*)d_in[2];
    const float* qgam   = (const float*)d_in[3];
    const float* qbet   = (const float*)d_in[4];
    const float* kgam   = (const float*)d_in[5];
    const float* kbet   = (const float*)d_in[6];
    float* out = (float*)d_out;

    const size_t NQKV = (size_t)BS * D_MODEL;       // 8M elems
    bf16* q_ws   = (bf16*)d_ws;                     // [B,H,S,Dh]
    bf16* k_ws   = q_ws + NQKV;                     // [B,H,S,Dh]
    bf16* v_ws   = k_ws + NQKV;                     // [B,H,Dh,S] (transposed)
    bf16* a_ws   = v_ws + NQKV;                     // attn out [B,S,D]
    bf16* xb_ws  = a_ws + NQKV;                     // x as bf16
    bf16* wqkvT  = xb_ws + NQKV;                    // [3072][1024]
    bf16* wprojT = wqkvT + (size_t)N3 * D_MODEL;    // [1024][1024]

    // 0) precision/layout pre-passes (one cheap sweep each)
    cvt_bf16_kernel<<<(unsigned)(NQKV / 2048), 256, 0, stream>>>(x, xb_ws);
    transpose_cvt_kernel<<<dim3(N3 / 32, D_MODEL / 32), 256, 0, stream>>>(
        w_qkv, wqkvT, D_MODEL, N3);
    transpose_cvt_kernel<<<dim3(D_MODEL / 32, D_MODEL / 32), 256, 0, stream>>>(
        w_proj, wprojT, D_MODEL, D_MODEL);

    // 1) QKV GEMM + fused QK LayerNorm (TDM double-buffered, padded LDS)
    qkv_ln_kernel<<<dim3(BS / 128, N3 / 128), 256, 0, stream>>>(
        xb_ws, wqkvT, qgam, qbet, kgam, kbet, q_ws, k_ws, v_ws);

    // 2) Flash attention per (b,h) (TDM double-buffered K / V^T, padded LDS)
    attn_kernel<<<dim3(SEQ / 256, 4 * HEADS), 256, 0, stream>>>(
        q_ws, k_ws, v_ws, a_ws);

    // 3) Output projection -> fp32 (TDM double-buffered, padded LDS)
    proj_kernel<<<dim3(BS / 128, D_MODEL / 128), 256, 0, stream>>>(
        a_ws, wprojT, out);
}
